// RSSM_1726576856303
// MI455X (gfx1250) — compile-verified, matrix operand order checked by reading, not run on hardware
//
#include <hip/hip_runtime.h>
#include <hip/hip_bf16.h>
#include <math.h>

// ---------------------------------------------------------------------------
// RSSM scan for MI455X (gfx1250, wave32, WMMA).
//  * bf16 weights pre-transposed to [N, Kpad] (L2-resident, ~20 MB)
//  * v_wmma_f32_16x16x32_bf16, f32 accumulate
//  * double-buffered LDS, A prefetched via registers, B streamed with
//    global_load_async_to_lds_b128 + s_wait_asynccnt (CDNA5 async path)
//  * 64 sequential steps, 9 kernels/step, graph-friendly
// ---------------------------------------------------------------------------

typedef __attribute__((ext_vector_type(16))) __bf16       v16bf;
typedef __attribute__((ext_vector_type(8)))  float        v8f;
typedef __attribute__((ext_vector_type(4)))  float        v4f;
typedef __attribute__((ext_vector_type(4)))  unsigned int v4u;

struct Q2 { v4u a, b; };   // 32 bytes == v16bf

#if defined(__gfx1250__)
#define GEMM_ASYNC_B 1
#else
#define GEMM_ASYNC_B 0
#endif

__device__ __forceinline__ unsigned short f2bf(float f) {
  union { float f; unsigned int u; } c; c.f = f;
  unsigned int u = c.u;
  u += 0x7FFFu + ((u >> 16) & 1u);          // round-to-nearest-even
  return (unsigned short)(u >> 16);
}

__device__ __forceinline__ float sigmoidf_(float x) { return 1.0f / (1.0f + expf(-x)); }
__device__ __forceinline__ float softplusf_(float x) { return x > 30.0f ? x : log1pf(expf(x)); }

// low 32 bits of a generic pointer to __shared__ == wave-relative LDS address
__device__ __forceinline__ unsigned lds_lo32(const void* p) {
  return (unsigned)(size_t)p;
}

__device__ __forceinline__ void async_load_b128(unsigned lds_off, const void* g) {
#if GEMM_ASYNC_B
  asm volatile("global_load_async_to_lds_b128 %0, %1, off"
               :: "v"(lds_off), "v"((unsigned long long)(size_t)g)
               : "memory");
#else
  (void)lds_off; (void)g;
#endif
}
__device__ __forceinline__ void wait_asynccnt0() {
#if GEMM_ASYNC_B
  asm volatile("s_wait_asynccnt 0x0" ::: "memory");
#endif
}

#define BM 128
#define BN 128
#define BK 32

// Stage one 16-float run of the (A0|A1|zeros) concatenated activation row.
template<bool ALIGNED>
__device__ __forceinline__ void stage_a_regs(
    float (&v)[16],
    const float* __restrict__ A0, int lda0, int K0,
    const float* __restrict__ A1, int lda1, int K1,
    int gm, int kk)
{
  if (ALIGNED) {            // K0 (and K0+K1) multiples of 16, no padding region
    const float* src = (kk < K0) ? (A0 + (size_t)gm * lda0 + kk)
                                 : (A1 + (size_t)gm * lda1 + (kk - K0));
    const v4f* p = (const v4f*)src;
    #pragma unroll
    for (int q = 0; q < 4; ++q) {
      v4f f = p[q];
      v[4*q+0]=f[0]; v[4*q+1]=f[1]; v[4*q+2]=f[2]; v[4*q+3]=f[3];
    }
  } else {                  // boundary / zero-pad (tiny inp GEMM only)
    #pragma unroll
    for (int e = 0; e < 16; ++e) {
      const int kg = kk + e;
      float x = 0.0f;
      if (kg < K0)           x = A0[(size_t)gm * lda0 + kg];
      else if (kg < K0 + K1) x = A1[(size_t)gm * lda1 + (kg - K0)];
      v[e] = x;
    }
  }
}

__device__ __forceinline__ void store_a_bf16(unsigned short* dst, const float (&v)[16]) {
  unsigned int pk[8];
  #pragma unroll
  for (int q = 0; q < 8; ++q)
    pk[q] = (unsigned int)f2bf(v[2*q]) | ((unsigned int)f2bf(v[2*q+1]) << 16);
  v4u* d = (v4u*)dst;
  d[0] = (v4u){pk[0], pk[1], pk[2], pk[3]};
  d[1] = (v4u){pk[4], pk[5], pk[6], pk[7]};
}

// ---------------------------------------------------------------------------
// C[M,N] = concatK(A0[M,K0], A1[M,K1]) * Wt^T + bias     (Wt: [N,Kpad] bf16)
// mode: 0 = bias only, 1 = bias + ELU
// 256 threads (8 waves), block tile 128x128x32, wave tile 64x32 (4x2 WMMA),
// double-buffered LDS, one barrier per K-step.
// ---------------------------------------------------------------------------
template<bool ALIGNED>
__global__ __launch_bounds__(256)
void k_gemm_bf16(const float* __restrict__ A0, int lda0, int K0,
                 const float* __restrict__ A1, int lda1, int K1,
                 const unsigned short* __restrict__ Wt, int Kpad,
                 const float* __restrict__ bias,
                 float* __restrict__ C, int N, int M, int mode)
{
  __shared__ __align__(16) unsigned short sA[2][BM * BK];
  __shared__ __align__(16) unsigned short sB[2][BN * BK];

  const int tid = threadIdx.x;
  const int m0  = blockIdx.y * BM;
  const int n0  = blockIdx.x * BN;

  const int waveId = tid >> 5;
  const int lane   = tid & 31;
  const int hi     = lane >> 4;            // lane half -> K sub-range
  const int lr     = lane & 15;            // row (A) / col (B) in fragment
  const int mw     = (waveId >> 2) * 64;   // wave M origin in tile
  const int nw     = (waveId & 3) * 32;    // wave N origin in tile

  v8f acc[4][2];
  #pragma unroll
  for (int i = 0; i < 4; ++i)
    #pragma unroll
    for (int j = 0; j < 2; ++j) acc[i][j] = {};

  // staging role: each thread owns a 16-element run of one row (A) / col (B)
  const int fillRow  = tid >> 1;           // 0..127
  const int fillHalf = tid & 1;            // k sub-run of 16
  const int gm_fill  = m0 + fillRow;
  const int gn_fill  = n0 + fillRow;
  const bool bIn     = (gn_fill < N);
  const unsigned short* bsrc = Wt + (size_t)gn_fill * Kpad + 16 * fillHalf;
  unsigned short* aSlot[2] = { &sA[0][fillRow * BK + 16 * fillHalf],
                               &sA[1][fillRow * BK + 16 * fillHalf] };
  unsigned short* bSlot[2] = { &sB[0][fillRow * BK + 16 * fillHalf],
                               &sB[1][fillRow * BK + 16 * fillHalf] };

  // ---- prologue: stage tile 0 ----
  float aregs[16];
  stage_a_regs<ALIGNED>(aregs, A0, lda0, K0, A1, lda1, K1, gm_fill, 16 * fillHalf);
#if GEMM_ASYNC_B
  if (bIn) {
    async_load_b128(lds_lo32(bSlot[0]),      bsrc);
    async_load_b128(lds_lo32(bSlot[0]) + 16, (const char*)bsrc + 16);
  } else {
    const v4u z = {0u,0u,0u,0u};
    ((v4u*)bSlot[0])[0] = z; ((v4u*)bSlot[0])[1] = z;
    ((v4u*)bSlot[1])[0] = z; ((v4u*)bSlot[1])[1] = z;
  }
#else
  v4u breg0 = {0u,0u,0u,0u}, breg1 = {0u,0u,0u,0u};
  if (bIn) { const v4u* g = (const v4u*)bsrc; breg0 = g[0]; breg1 = g[1]; }
#endif

  for (int k0 = 0; k0 < Kpad; k0 += BK) {
    const int cur = (k0 >> 5) & 1;

    // publish staged tile `cur`
    store_a_bf16(aSlot[cur], aregs);
#if GEMM_ASYNC_B
    wait_asynccnt0();                       // B tile `cur` landed in LDS
#else
    { v4u* d = (v4u*)bSlot[cur]; d[0] = breg0; d[1] = breg1; }
#endif
    __syncthreads();

    // prefetch tile cur^1 (overlaps with WMMA below)
    const int kn = k0 + BK;
    if (kn < Kpad) {
      stage_a_regs<ALIGNED>(aregs, A0, lda0, K0, A1, lda1, K1,
                            gm_fill, kn + 16 * fillHalf);
#if GEMM_ASYNC_B
      if (bIn) {
        const unsigned short* g = bsrc + kn;
        async_load_b128(lds_lo32(bSlot[cur ^ 1]),      g);
        async_load_b128(lds_lo32(bSlot[cur ^ 1]) + 16, (const char*)g + 16);
      }
#else
      if (bIn) { const v4u* g = (const v4u*)(bsrc + kn); breg0 = g[0]; breg1 = g[1]; }
#endif
    }

    // fragment loads (ISA layout: K = 16*(e/8) + 8*(lane/16) + e%8) + WMMA
    v16bf afrag[4], bfrag[2];
    #pragma unroll
    for (int mi = 0; mi < 4; ++mi) {
      const int row = mw + mi * 16 + lr;
      const v4u* p = (const v4u*)&sA[cur][row * BK + 8 * hi];
      Q2 qq; qq.a = p[0]; qq.b = p[2];
      afrag[mi] = __builtin_bit_cast(v16bf, qq);
    }
    #pragma unroll
    for (int ni = 0; ni < 2; ++ni) {
      const int col = nw + ni * 16 + lr;
      const v4u* p = (const v4u*)&sB[cur][col * BK + 8 * hi];
      Q2 qq; qq.a = p[0]; qq.b = p[2];
      bfrag[ni] = __builtin_bit_cast(v16bf, qq);
    }
    #pragma unroll
    for (int mi = 0; mi < 4; ++mi)
      #pragma unroll
      for (int ni = 0; ni < 2; ++ni)
        acc[mi][ni] = __builtin_amdgcn_wmma_f32_16x16x32_bf16(
            false, afrag[mi], false, bfrag[ni], (short)0, acc[mi][ni], false, false);
  }

  // ---- epilogue: C/D layout lane L -> N = L%16, M = vgpr + 8*(L/16) ----
  #pragma unroll
  for (int mi = 0; mi < 4; ++mi) {
    #pragma unroll
    for (int ni = 0; ni < 2; ++ni) {
      const int col = n0 + nw + ni * 16 + lr;
      if (col >= N) continue;
      const float bv = bias ? bias[col] : 0.0f;
      #pragma unroll
      for (int v = 0; v < 8; ++v) {
        const int row = m0 + mw + mi * 16 + hi * 8 + v;
        if (row >= M) continue;
        float x = acc[mi][ni][v] + bv;
        if (mode == 1) x = x > 0.0f ? x : expm1f(x);   // ELU(alpha=1)
        C[(size_t)row * N + col] = x;
      }
    }
  }
}

// ---------------------------------------------------------------------------
// Weight prep: W[K,N] f32 row-major -> Wt[N,Kpad] bf16 (transposed, zero-pad)
// ---------------------------------------------------------------------------
__global__ void k_transpose_bf16(const float* __restrict__ W,
                                 unsigned short* __restrict__ Wt,
                                 int K, int N, int Kpad)
{
  const int idx = blockIdx.x * blockDim.x + threadIdx.x;
  const int n = idx / Kpad, k = idx % Kpad;
  if (n >= N) return;
  Wt[(size_t)n * Kpad + k] = (k < K) ? f2bf(W[(size_t)k * N + n]) : (unsigned short)0;
}

__global__ void k_zero(float* __restrict__ p, int n) {
  const int i = blockIdx.x * blockDim.x + threadIdx.x;
  if (i < n) p[i] = 0.0f;
}

// ---------------------------------------------------------------------------
// GRU gating: parts[B,3072] + deter_old -> deter_new; scatter deter into out
// out layout: [2, B, T, 1120], deter slice at offset 96
// ---------------------------------------------------------------------------
__global__ void k_gru_gate(const float* __restrict__ parts,
                           const float* __restrict__ deter_old,
                           float* __restrict__ deter_new,
                           float* __restrict__ out, int t)
{
  const int idx = blockIdx.x * blockDim.x + threadIdx.x;   // 512*1024
  const int b = idx >> 10, j = idx & 1023;
  const float* p = parts + (size_t)b * 3072;
  const float r = sigmoidf_(p[j]);
  const float c = tanhf(r * p[j + 1024]);
  const float u = sigmoidf_(p[j + 2048] - 1.0f);           // update_bias = -1
  const float dn = u * c + (1.0f - u) * deter_old[idx];
  deter_new[idx] = dn;
  const size_t o0 = (((size_t)0 * 512 + b) * 64 + t) * 1120 + 96 + j;  // post
  const size_t o1 = (((size_t)1 * 512 + b) * 64 + t) * 1120 + 96 + j;  // prior
  out[o0] = dn;
  out[o1] = dn;
}

// ---------------------------------------------------------------------------
// Stats epilogue: stats[B,64] -> mean/std/sample into out[sIdx]; optional carry
// ---------------------------------------------------------------------------
__global__ void k_stats(const float* __restrict__ st,
                        const float* __restrict__ eps,   // [T,B,32]
                        float* __restrict__ out, int t, int sIdx,
                        float* __restrict__ stoch_carry)
{
  const int idx = blockIdx.x * blockDim.x + threadIdx.x;   // 512*32
  const int b = idx >> 5, j = idx & 31;
  const float mean = st[b * 64 + j];
  const float stdv = softplusf_(st[b * 64 + 32 + j]) + 0.1f;
  const float smpl = mean + stdv * eps[((size_t)t * 512 + b) * 32 + j];
  const size_t base = (((size_t)sIdx * 512 + b) * 64 + t) * 1120;
  out[base + j]      = mean;
  out[base + 32 + j] = stdv;
  out[base + 64 + j] = smpl;
  if (stoch_carry) stoch_carry[b * 32 + j] = smpl;
}

// ---------------------------------------------------------------------------
extern "C" void kernel_launch(void* const* d_in, const int* in_sizes, int n_in,
                              void* d_out, int out_size, void* d_ws, size_t ws_size,
                              hipStream_t stream)
{
  (void)in_sizes; (void)n_in; (void)out_size; (void)ws_size;
  const float* embed  = (const float*)d_in[0];   // [512,64,1536]
  const float* action = (const float*)d_in[1];   // [512,64,12]
  const float* eps_pr = (const float*)d_in[2];   // [64,512,32]
  const float* eps_po = (const float*)d_in[3];
  const float* w_inp  = (const float*)d_in[4];   const float* b_inp  = (const float*)d_in[5];
  const float* w_gru  = (const float*)d_in[6];   const float* b_gru  = (const float*)d_in[7];
  const float* w_img  = (const float*)d_in[8];   const float* b_img  = (const float*)d_in[9];
  const float* w_obs  = (const float*)d_in[10];  const float* b_obs  = (const float*)d_in[11];
  const float* w_ims  = (const float*)d_in[12];  const float* b_ims  = (const float*)d_in[13];
  const float* w_obss = (const float*)d_in[14];  const float* b_obss = (const float*)d_in[15];
  float* out = (float*)d_out;

  char* ws = (char*)d_ws;
  size_t off = 0;
  auto alloc = [&](size_t bytes) -> void* {
    off = (off + 255) & ~(size_t)255;
    void* p = ws + off; off += bytes; return p;
  };

  // bf16 transposed weights (L2-resident working set, ~20 MB)
  unsigned short* wt_inp  = (unsigned short*)alloc((size_t)1024 * 64   * 2);
  unsigned short* wt_gru  = (unsigned short*)alloc((size_t)3072 * 2048 * 2);
  unsigned short* wt_img  = (unsigned short*)alloc((size_t)1024 * 1024 * 2);
  unsigned short* wt_obs  = (unsigned short*)alloc((size_t)1024 * 2560 * 2);
  unsigned short* wt_ims  = (unsigned short*)alloc((size_t)64   * 1024 * 2);
  unsigned short* wt_obss = (unsigned short*)alloc((size_t)64   * 1024 * 2);
  // activations (f32)
  float* x      = (float*)alloc((size_t)512 * 1024 * 4);
  float* parts  = (float*)alloc((size_t)512 * 3072 * 4);
  float* deter0 = (float*)alloc((size_t)512 * 1024 * 4);
  float* deter1 = (float*)alloc((size_t)512 * 1024 * 4);
  float* hbuf   = (float*)alloc((size_t)512 * 1024 * 4);
  float* hobuf  = (float*)alloc((size_t)512 * 1024 * 4);
  float* stp    = (float*)alloc((size_t)512 * 64 * 4);
  float* sto    = (float*)alloc((size_t)512 * 64 * 4);
  float* stoch  = (float*)alloc((size_t)512 * 32 * 4);

  auto tgrid = [](int n, int kpad) { return dim3((unsigned)((n * kpad + 255) / 256)); };
  k_transpose_bf16<<<tgrid(1024, 64),   256, 0, stream>>>(w_inp,  wt_inp,  44,   1024, 64);
  k_transpose_bf16<<<tgrid(3072, 2048), 256, 0, stream>>>(w_gru,  wt_gru,  2048, 3072, 2048);
  k_transpose_bf16<<<tgrid(1024, 1024), 256, 0, stream>>>(w_img,  wt_img,  1024, 1024, 1024);
  k_transpose_bf16<<<tgrid(1024, 2560), 256, 0, stream>>>(w_obs,  wt_obs,  2560, 1024, 2560);
  k_transpose_bf16<<<tgrid(64, 1024),   256, 0, stream>>>(w_ims,  wt_ims,  1024, 64,   1024);
  k_transpose_bf16<<<tgrid(64, 1024),   256, 0, stream>>>(w_obss, wt_obss, 1024, 64,   1024);

  k_zero<<<64,   256, 0, stream>>>(stoch,  512 * 32);
  k_zero<<<2048, 256, 0, stream>>>(deter0, 512 * 1024);

  for (int t = 0; t < 64; ++t) {
    float* dcur = (t & 1) ? deter1 : deter0;
    float* dnxt = (t & 1) ? deter0 : deter1;

    // x = elu([stoch(32)|act(12)] @ w_inp + b_inp)           [512,1024]
    k_gemm_bf16<false><<<dim3(8, 4), 256, 0, stream>>>(
        stoch, 32, 32, action + (size_t)t * 12, 64 * 12, 12,
        wt_inp, 64, b_inp, x, 1024, 512, 1);

    // parts = [x|deter] @ w_gru + b_gru                      [512,3072]
    k_gemm_bf16<true><<<dim3(24, 4), 256, 0, stream>>>(
        x, 1024, 1024, dcur, 1024, 1024,
        wt_gru, 2048, b_gru, parts, 3072, 512, 0);

    // GRU gating -> deter_new (+ scatter deter into out)
    k_gru_gate<<<2048, 256, 0, stream>>>(parts, dcur, dnxt, out, t);

    // h = elu(deter_new @ w_img + b_img)                     [512,1024]
    k_gemm_bf16<true><<<dim3(8, 4), 256, 0, stream>>>(
        dnxt, 1024, 1024, nullptr, 0, 0,
        wt_img, 1024, b_img, hbuf, 1024, 512, 1);

    // prior stats                                            [512,64]
    k_gemm_bf16<true><<<dim3(1, 4), 256, 0, stream>>>(
        hbuf, 1024, 1024, nullptr, 0, 0,
        wt_ims, 1024, b_ims, stp, 64, 512, 0);
    k_stats<<<64, 256, 0, stream>>>(stp, eps_pr, out, t, 1, nullptr);

    // ho = elu([deter_new|embed_t] @ w_obs + b_obs)          [512,1024]
    k_gemm_bf16<true><<<dim3(8, 4), 256, 0, stream>>>(
        dnxt, 1024, 1024, embed + (size_t)t * 1536, 64 * 1536, 1536,
        wt_obs, 2560, b_obs, hobuf, 1024, 512, 1);

    // posterior stats + stoch carry                          [512,64]
    k_gemm_bf16<true><<<dim3(1, 4), 256, 0, stream>>>(
        hobuf, 1024, 1024, nullptr, 0, 0,
        wt_obss, 1024, b_obss, sto, 64, 512, 0);
    k_stats<<<64, 256, 0, stream>>>(sto, eps_po, out, t, 0, stoch);
  }
}
